// DecoderWithAttention_87617332838414
// MI455X (gfx1250) — compile-verified, matrix operand order checked by reading, not run on hardware
//
#include <hip/hip_runtime.h>
#include <hip/hip_bf16.h>
#include <math.h>

// ---------------- problem constants ----------------
#define B_      64
#define P_      196
#define ENC_    2048
#define DEC_    512
#define EMB_    512
#define ATT_    1024
#define ATTC_   1024
#define VOCAB_  10000
#define MAXLEN_ 22
#define T_      21
#define K2_     1732          // ATT + DEC + P
#define K2PAD_  1760          // padded to multiple of 32

typedef __attribute__((ext_vector_type(16))) _Float16 v16h;
typedef __attribute__((ext_vector_type(8)))  _Float16 v8h;
typedef __attribute__((ext_vector_type(8)))  float    v8f;

// ---------------- small device helpers ----------------
__device__ inline float sigm(float x) { return 1.0f / (1.0f + __expf(-x)); }

__device__ inline float block_reduce_sum(float v) {
  __shared__ float s[256];
  int tid = threadIdx.x;
  s[tid] = v; __syncthreads();
  for (int off = 128; off > 0; off >>= 1) {
    if (tid < off) s[tid] += s[tid + off];
    __syncthreads();
  }
  float r = s[0]; __syncthreads();
  return r;
}

__device__ inline float block_reduce_max(float v) {
  __shared__ float s[256];
  int tid = threadIdx.x;
  s[tid] = v; __syncthreads();
  for (int off = 128; off > 0; off >>= 1) {
    if (tid < off) s[tid] = fmaxf(s[tid], s[tid + off]);
    __syncthreads();
  }
  float r = s[0]; __syncthreads();
  return r;
}

// Load a 16x32 f16 fragment (A or B side; both are row-major RxK here since
// C = A * B^T).  Per CDNA5 ISA layout: lane L (r = L&15, h = L>>4) holds
// K = [h*8, h*8+8) in v[0..7] and K = [16+h*8, 16+h*8+8) in v[8..15].
// Each run is 16 contiguous bytes -> two b128 loads.
__device__ inline v16h load_frag16(const _Float16* __restrict__ base, int ld,
                                   int row, int k0, int lane) {
  int r = lane & 15, h = lane >> 4;
  const _Float16* p0 = base + (size_t)(row + r) * ld + k0 + h * 8;
  v8h a = *(const v8h*)p0;
  v8h b = *(const v8h*)(p0 + 16);
  v16h o;
#pragma unroll
  for (int i = 0; i < 8; ++i) { o[i] = a[i]; o[i + 8] = b[i]; }
  return o;
}

// ---------------- WMMA GEMM main: C[M,N] = A[M,K] * B[N,K]^T (+bias)(+=C)(*rowmask)
// 128 threads = 4 waves.  Each wave owns a FIXED 16x64 output strip (4
// accumulators, no inner guards -> 4 back-to-back v_wmma per K-step sharing one
// A fragment; B loads hide the WMMA->WMMA hazard).  Block tile = 16 x 256.
// Must be launched with N a multiple of 64 (caller splits off the edge).
__global__ void k_gemm_wmma4(const _Float16* __restrict__ A,
                             const _Float16* __restrict__ Bm,
                             const float* __restrict__ bias,
                             float* __restrict__ C,
                             const float* __restrict__ rowmask,
                             int M, int N, int K, int ldc, int accumulate) {
  int lane = threadIdx.x & 31;
  int wave = threadIdx.x >> 5;
  int tileM  = blockIdx.y * 16;
  int tileN0 = blockIdx.x * 256 + wave * 64;
  if (tileM >= M || tileN0 >= N) return;   // wave-uniform exit only

  v8f a0 = {}, a1 = {}, a2 = {}, a3 = {};
  for (int k0 = 0; k0 < K; k0 += 32) {
    v16h af = load_frag16(A, K, tileM, k0, lane);
    if (k0 + 32 < K)   // hint next A K-slab into cache (global_prefetch_b8)
      __builtin_prefetch(A + (size_t)(tileM + (lane & 15)) * K + k0 + 32, 0, 0);
    v16h b0 = load_frag16(Bm, K, tileN0 +  0, k0, lane);
    v16h b1 = load_frag16(Bm, K, tileN0 + 16, k0, lane);
    v16h b2 = load_frag16(Bm, K, tileN0 + 32, k0, lane);
    v16h b3 = load_frag16(Bm, K, tileN0 + 48, k0, lane);
    a0 = __builtin_amdgcn_wmma_f32_16x16x32_f16(false, af, false, b0, (short)0, a0, false, false);
    a1 = __builtin_amdgcn_wmma_f32_16x16x32_f16(false, af, false, b1, (short)0, a1, false, false);
    a2 = __builtin_amdgcn_wmma_f32_16x16x32_f16(false, af, false, b2, (short)0, a2, false, false);
    a3 = __builtin_amdgcn_wmma_f32_16x16x32_f16(false, af, false, b3, (short)0, a3, false, false);
  }

  int nn = lane & 15;
  int mbase = tileM + ((lane >> 4) ? 8 : 0);
  v8f accs[4] = {a0, a1, a2, a3};
#pragma unroll
  for (int ti = 0; ti < 4; ++ti) {
    int n = tileN0 + ti * 16 + nn;
    float bn = bias ? bias[n] : 0.0f;
#pragma unroll
    for (int v = 0; v < 8; ++v) {
      int m = mbase + v;
      float r = accs[ti][v] + bn;
      float* cp = C + (size_t)m * ldc + n;
      if (accumulate) r += *cp;
      if (rowmask)    r *= rowmask[m];
      *cp = r;
    }
  }
}

// Edge kernel: one 16x16 tile per wave (handles N % 64 remainder columns).
__global__ void k_gemm_wmma1(const _Float16* __restrict__ A,
                             const _Float16* __restrict__ Bm,
                             const float* __restrict__ bias,
                             float* __restrict__ C,
                             const float* __restrict__ rowmask,
                             int M, int N, int K, int ldc, int accumulate) {
  int lane = threadIdx.x & 31;
  int wave = threadIdx.x >> 5;
  int tileM = blockIdx.y * 16;
  int tileN = blockIdx.x * 64 + wave * 16;
  if (tileM >= M || tileN >= N) return;   // wave-uniform

  v8f acc = {};
  for (int k0 = 0; k0 < K; k0 += 32) {
    v16h a = load_frag16(A,  K, tileM, k0, lane);
    v16h b = load_frag16(Bm, K, tileN, k0, lane);
    acc = __builtin_amdgcn_wmma_f32_16x16x32_f16(false, a, false, b, (short)0, acc, false, false);
  }
  int n = tileN + (lane & 15);
  int mbase = tileM + ((lane >> 4) ? 8 : 0);
  float bn = bias ? bias[n] : 0.0f;
#pragma unroll
  for (int v = 0; v < 8; ++v) {
    int m = mbase + v;
    float r = acc[v] + bn;
    float* cp = C + (size_t)m * ldc + n;
    if (accumulate) r += *cp;
    if (rowmask)    r *= rowmask[m];
    *cp = r;
  }
}

// ---------------- setup kernels ----------------
// stable argsort of -lens (desc by len, ties by original index)
__global__ void k_sort(const int* __restrict__ lens2, const int* __restrict__ caps,
                       int* sort_ind, int* dec_len, int* caps_s,
                       float* out_caps, float* out_declen, float* out_sortind) {
  __shared__ int si[B_];
  int i = threadIdx.x;
  int li = lens2[i];      // caption_lengths[i,0]
  int rank = 0;
  for (int j = 0; j < B_; ++j) {
    int lj = lens2[j];
    if (lj > li || (lj == li && j < i)) rank++;
  }
  si[rank] = i;
  __syncthreads();
  int s = si[i];
  sort_ind[i] = s;
  dec_len[i] = lens2[s] - 1;
  out_declen[i]  = (float)(lens2[s] - 1);
  out_sortind[i] = (float)s;
  for (int t = 0; t < MAXLEN_; ++t) {
    int tok = caps[s * MAXLEN_ + t];
    caps_s[i * MAXLEN_ + t] = tok;
    out_caps[i * MAXLEN_ + t] = (float)tok;
  }
}

// gather enc by sort_ind and cast: enc_s f32, enc_h f16.  grid = B*P blocks.
__global__ void k_gather_cast(const float* __restrict__ enc_in,
                              const int* __restrict__ sort_ind,
                              float* __restrict__ enc_s, _Float16* __restrict__ enc_h) {
  int row = blockIdx.x;          // b*P + p
  int b = row / P_, p = row - b * P_;
  int sb = sort_ind[b];
  const float* src = enc_in + ((size_t)sb * P_ + p) * ENC_;
  float* d32 = enc_s + (size_t)row * ENC_;
  _Float16* d16 = enc_h + (size_t)row * ENC_;
  for (int e = threadIdx.x; e < ENC_; e += 256) {
    float v = src[e];
    d32[e] = v; d16[e] = (_Float16)v;
  }
}

__global__ void k_mean(const float* __restrict__ enc_s,
                       float* __restrict__ mean_f, _Float16* __restrict__ mean_h) {
  int b = blockIdx.x;
  for (int e = threadIdx.x; e < ENC_; e += 256) {
    float s = 0.0f;
    const float* base = enc_s + (size_t)b * P_ * ENC_ + e;
    for (int p = 0; p < P_; ++p) s += base[(size_t)p * ENC_];
    s *= (1.0f / (float)P_);
    mean_f[b * ENC_ + e] = s;
    mean_h[b * ENC_ + e] = (_Float16)s;
  }
}

__global__ void k_cast(const float* __restrict__ src, _Float16* __restrict__ dst, int n) {
  int i = blockIdx.x * 256 + threadIdx.x;
  if (i < n) dst[i] = (_Float16)src[i];
}

__global__ void k_cast_pad(const float* __restrict__ src, _Float16* __restrict__ dst) {
  // lstm2_Wih: (2048, 1732) -> (2048, 1760) zero-padded
  int i = blockIdx.x * 256 + threadIdx.x;
  if (i >= 2048 * K2PAD_) return;
  int r = i / K2PAD_, c = i - r * K2PAD_;
  dst[i] = (c < K2_) ? (_Float16)src[r * K2_ + c] : (_Float16)0.0f;
}

__global__ void k_init_hc(const float* __restrict__ h, const float* __restrict__ c,
                          float* h2, float* c2, _Float16* h_h, _Float16* h2_h) {
  int i = blockIdx.x * 256 + threadIdx.x;
  if (i >= B_ * DEC_) return;
  float hv = h[i];
  h2[i] = hv; c2[i] = c[i];
  h_h[i] = (_Float16)hv; h2_h[i] = (_Float16)hv;
}

// ---------------- per-step kernels ----------------
__global__ void k_active(const int* __restrict__ dec_len, float* amf, int t) {
  int i = threadIdx.x;
  if (i < B_) amf[i] = (t < dec_len[i]) ? 1.0f : 0.0f;
}

// e[b,p] = sum_a relu(att1[b,p,a] + att2[b,a]) * wa[a] + b0
__global__ void k_e_dot(const float* __restrict__ att1, const float* __restrict__ att2,
                        const float* __restrict__ wa, const float* __restrict__ b0,
                        float* __restrict__ e_buf) {
  int row = blockIdx.x;                 // b*P + p
  int b = row / P_;
  const float* a1 = att1 + (size_t)row * ATT_;
  const float* a2 = att2 + (size_t)b * ATT_;
  float s = 0.0f;
  for (int a = threadIdx.x; a < ATT_; a += 256)
    s += fmaxf(a1[a] + a2[a], 0.0f) * wa[a];
  s = block_reduce_sum(s);
  if (threadIdx.x == 0) e_buf[row] = s + b0[0];
}

__global__ void k_softmax_p(const float* __restrict__ e_buf, float* __restrict__ alpha,
                            float* __restrict__ out_alphas, const float* __restrict__ amf,
                            int t) {
  int b = blockIdx.x, tid = threadIdx.x;
  float v = (tid < P_) ? e_buf[b * P_ + tid] : -3.0e38f;
  float mx = block_reduce_max(v);
  float ex = (tid < P_) ? __expf(v - mx) : 0.0f;
  float sm = block_reduce_sum(ex);
  if (tid < P_) {
    float a = ex / sm;
    alpha[b * P_ + tid] = a;
    out_alphas[((size_t)b * T_ + t) * P_ + tid] = a * amf[b];
  }
}

// awe[b,e] = sum_p enc[b,p,e]*alpha[b,p]; cast f16 too
__global__ void k_awe(const float* __restrict__ enc_s, const float* __restrict__ alpha,
                      float* __restrict__ awe, _Float16* __restrict__ awe_h) {
  int b = blockIdx.x, e = blockIdx.y * 256 + threadIdx.x;
  const float* base = enc_s + (size_t)b * P_ * ENC_ + e;
  const float* al = alpha + b * P_;
  float s = 0.0f;
  for (int p = 0; p < P_; ++p) s += base[(size_t)p * ENC_] * al[p];
  awe[b * ENC_ + e] = s;
  awe_h[b * ENC_ + e] = (_Float16)s;
}

// ec[b,a] = sum_p relu(catt1[b,p,a]+catt2[b,a]) * wp[p] + c0
__global__ void k_ec(const float* __restrict__ catt1, const float* __restrict__ catt2,
                     const float* __restrict__ wp, const float* __restrict__ c0,
                     float* __restrict__ ec) {
  int idx = blockIdx.x * 256 + threadIdx.x;      // b*ATTC + a
  if (idx >= B_ * ATTC_) return;
  int b = idx >> 10;
  int a = idx & (ATTC_ - 1);
  float c2v = catt2[idx];
  const float* base = catt1 + (size_t)b * P_ * ATTC_ + a;
  float s = 0.0f;
  for (int p = 0; p < P_; ++p)
    s += fmaxf(base[(size_t)p * ATTC_] + c2v, 0.0f) * wp[p];
  ec[idx] = s + c0[0];
}

__global__ void k_softmax_a(const float* __restrict__ ec, float* __restrict__ alpha_c,
                            float* __restrict__ out_ac, const float* __restrict__ amf,
                            int t) {
  int b = blockIdx.x, tid = threadIdx.x;
  float loc[4]; float mx = -3.0e38f;
#pragma unroll
  for (int i = 0; i < 4; ++i) { loc[i] = ec[b * ATTC_ + tid + i * 256]; mx = fmaxf(mx, loc[i]); }
  mx = block_reduce_max(mx);
  float s = 0.0f;
#pragma unroll
  for (int i = 0; i < 4; ++i) { loc[i] = __expf(loc[i] - mx); s += loc[i]; }
  s = block_reduce_sum(s);
  float m = amf[b];
#pragma unroll
  for (int i = 0; i < 4; ++i) {
    int j = tid + i * 256;
    float a = loc[i] / s;
    alpha_c[b * ATTC_ + j] = a;
    out_ac[((size_t)b * T_ + t) * ATTC_ + j] = a * m;
  }
}

// awce[b,p] = sum_a imf[b,p,a]*alpha_c[b,a]
__global__ void k_awce(const float* __restrict__ imf, const float* __restrict__ alpha_c,
                       float* __restrict__ awce) {
  int row = blockIdx.x;             // b*P + p
  int b = row / P_;
  const float* base = imf + (size_t)row * ATTC_;
  const float* al = alpha_c + b * ATTC_;
  float s = 0.0f;
  for (int a = threadIdx.x; a < ATTC_; a += 256) s += base[a] * al[a];
  s = block_reduce_sum(s);
  if (threadIdx.x == 0) awce[row] = s;
}

// coff = r_std / (r_std + c_std); masked std over awe2 (64x1024) and awce (64x196)
__global__ void k_std_coff(const float* __restrict__ awe2, const float* __restrict__ awce,
                           const float* __restrict__ amf, float* __restrict__ scal) {
  int tid = threadIdx.x;
  float cm = (tid < B_) ? amf[tid] : 0.0f;
  cm = block_reduce_sum(cm);
  float s1 = 0.0f, s2 = 0.0f;
  for (int i = tid; i < B_ * ATT_; i += 256) {
    float m = amf[i >> 10], x = awe2[i];
    s1 += x * m; s2 += x * x * m;
  }
  s1 = block_reduce_sum(s1); s2 = block_reduce_sum(s2);
  float t1 = 0.0f, t2 = 0.0f;
  for (int i = tid; i < B_ * P_; i += 256) {
    float m = amf[i / P_], x = awce[i];
    t1 += x * m; t2 += x * x * m;
  }
  t1 = block_reduce_sum(t1); t2 = block_reduce_sum(t2);
  if (tid == 0) {
    float n1 = cm * (float)ATT_;
    float var1 = (s2 - s1 * s1 / n1) / (n1 - 1.0f);
    float n2 = cm * (float)P_;
    float var2 = (t2 - t1 * t1 / n2) / (n2 - 1.0f);
    float rs = sqrtf(fmaxf(var1, 0.0f)), cs = sqrtf(fmaxf(var2, 0.0f));
    scal[0] = rs / (rs + cs);
  }
}

// x1 = concat(embedding[caps_s[:,t]], h2, img_gm) -> f16 (64,2048)
__global__ void k_build_x1(const float* __restrict__ embedding, const int* __restrict__ caps_s,
                           const float* __restrict__ h2, const float* __restrict__ img_gm,
                           _Float16* __restrict__ x1_h, int t) {
  int b = blockIdx.x;
  int tok = caps_s[b * MAXLEN_ + t];
  for (int j = threadIdx.x; j < 2048; j += 256) {
    float v;
    if (j < EMB_)             v = embedding[(size_t)tok * EMB_ + j];
    else if (j < EMB_ + DEC_) v = h2[b * DEC_ + (j - EMB_)];
    else                      v = img_gm[b * ATT_ + (j - EMB_ - DEC_)];
    x1_h[(size_t)b * 2048 + j] = (_Float16)v;
  }
}

// x2 = concat((1-coff)*awce, coff*awe2, h, zero-pad) -> f16 (64,1760)
__global__ void k_build_x2(const float* __restrict__ awce, const float* __restrict__ awe2,
                           const float* __restrict__ h, const float* __restrict__ scal,
                           _Float16* __restrict__ x2_h) {
  int b = blockIdx.x;
  float coff = scal[0];
  for (int j = threadIdx.x; j < K2PAD_; j += 256) {
    float v;
    if (j < P_)             v = (1.0f - coff) * awce[b * P_ + j];
    else if (j < P_ + ATT_) v = coff * awe2[b * ATT_ + (j - P_)];
    else if (j < K2_)       v = h[b * DEC_ + (j - P_ - ATT_)];
    else                    v = 0.0f;
    x2_h[(size_t)b * K2PAD_ + j] = (_Float16)v;
  }
}

// gated LSTM update with mask; writes f32 state + f16 copy of h
__global__ void k_lstm(const float* __restrict__ g, float* __restrict__ h,
                       float* __restrict__ c, _Float16* __restrict__ h_h,
                       const float* __restrict__ amf) {
  int idx = blockIdx.x * 256 + threadIdx.x;
  if (idx >= B_ * DEC_) return;
  int b = idx >> 9, j = idx & (DEC_ - 1);
  const float* gb = g + (size_t)b * (4 * DEC_);
  float i_ = gb[j], f_ = gb[DEC_ + j], gg = gb[2 * DEC_ + j], o_ = gb[3 * DEC_ + j];
  float cn = sigm(f_) * c[idx] + sigm(i_) * tanhf(gg);
  float hn = sigm(o_) * tanhf(cn);
  bool act = amf[b] > 0.5f;
  float hv = act ? hn : h[idx];
  float cv = act ? cn : c[idx];
  h[idx] = hv; c[idx] = cv; h_h[idx] = (_Float16)hv;
}

// ---------------- host orchestration ----------------
static inline void gemm(hipStream_t s, const _Float16* A, const _Float16* Bm,
                        const float* bias, float* C, const float* rowmask,
                        int M, int N, int K, int ldc, int acc) {
  int N64 = N & ~63;          // bulk handled by guard-free 4-tile kernel
  if (N64 > 0) {
    dim3 g((N64 + 255) / 256, M / 16);
    k_gemm_wmma4<<<g, 128, 0, s>>>(A, Bm, bias, C, rowmask, M, N64, K, ldc, acc);
  }
  if (N > N64) {              // ragged edge (only the vocab GEMM hits this)
    dim3 g(((N - N64) + 63) / 64, M / 16);
    k_gemm_wmma1<<<g, 128, 0, s>>>(A, Bm + (size_t)N64 * K,
                                   bias ? bias + N64 : nullptr,
                                   C + N64, rowmask, M, N - N64, K, ldc, acc);
  }
}

extern "C" void kernel_launch(void* const* d_in, const int* in_sizes, int n_in,
                              void* d_out, int out_size, void* d_ws, size_t ws_size,
                              hipStream_t stream) {
  (void)in_sizes; (void)n_in; (void)out_size; (void)ws_size;
  // -------- inputs --------
  const float* enc_in    = (const float*)d_in[0];
  const int*   caps_in   = (const int*)d_in[1];
  const int*   lens_in   = (const int*)d_in[2];
  const float* att_enc_W = (const float*)d_in[3];  const float* att_enc_b = (const float*)d_in[4];
  const float* att_dec_W = (const float*)d_in[5];  const float* att_dec_b = (const float*)d_in[6];
  const float* att_full_W= (const float*)d_in[7];  const float* att_full_b= (const float*)d_in[8];
  const float* chan_enc_W= (const float*)d_in[9];  const float* chan_enc_b= (const float*)d_in[10];
  const float* chan_dec_W= (const float*)d_in[11]; const float* chan_dec_b= (const float*)d_in[12];
  const float* chan_out_W= (const float*)d_in[13]; const float* chan_out_b= (const float*)d_in[14];
  const float* chan_full_W=(const float*)d_in[15]; const float* chan_full_b=(const float*)d_in[16];
  const float* embedding = (const float*)d_in[17];
  const float* l1_Wih = (const float*)d_in[18]; const float* l1_bih = (const float*)d_in[19];
  const float* l1_Whh = (const float*)d_in[20]; const float* l1_bhh = (const float*)d_in[21];
  const float* l2_Wih = (const float*)d_in[22]; const float* l2_bih = (const float*)d_in[23];
  const float* l2_Whh = (const float*)d_in[24]; const float* l2_bhh = (const float*)d_in[25];
  const float* init_h_W = (const float*)d_in[26]; const float* init_h_b = (const float*)d_in[27];
  const float* init_c_W = (const float*)d_in[28]; const float* init_c_b = (const float*)d_in[29];
  const float* att_div_W = (const float*)d_in[30]; const float* att_div_b = (const float*)d_in[31];
  const float* feat_div_W= (const float*)d_in[32]; const float* feat_div_b= (const float*)d_in[33];
  const float* fc_W = (const float*)d_in[34]; const float* fc_b = (const float*)d_in[35];

  float* outf = (float*)d_out;
  // output layout (flat f32, reference tuple order)
  const size_t OUT_PRED    = 0;
  const size_t OUT_CAPS    = OUT_PRED + (size_t)B_ * T_ * VOCAB_;
  const size_t OUT_DECLEN  = OUT_CAPS + (size_t)B_ * MAXLEN_;
  const size_t OUT_ALPHAS  = OUT_DECLEN + B_;
  const size_t OUT_ALPHASC = OUT_ALPHAS + (size_t)B_ * T_ * P_;
  const size_t OUT_SORTIND = OUT_ALPHASC + (size_t)B_ * T_ * ATTC_;

  // -------- workspace bump allocator (deterministic) --------
  char* wp = (char*)d_ws;
  auto alloc = [&](size_t bytes) -> void* {
    void* p = (void*)wp;
    wp += (bytes + 255) & ~(size_t)255;
    return p;
  };
  float* enc_s  = (float*)alloc((size_t)B_ * P_ * ENC_ * 4);
  _Float16* enc_h = (_Float16*)alloc((size_t)B_ * P_ * ENC_ * 2);
  float* att1 = (float*)alloc((size_t)B_ * P_ * ATT_ * 4);
  float* catt1= (float*)alloc((size_t)B_ * P_ * ATTC_ * 4);
  float* imf  = (float*)alloc((size_t)B_ * P_ * ATTC_ * 4);
  float* mean_f = (float*)alloc((size_t)B_ * ENC_ * 4);
  _Float16* mean_h = (_Float16*)alloc((size_t)B_ * ENC_ * 2);
  float* img_gm = (float*)alloc((size_t)B_ * ATT_ * 4);
  float *h  = (float*)alloc(B_ * DEC_ * 4), *c  = (float*)alloc(B_ * DEC_ * 4);
  float *h2 = (float*)alloc(B_ * DEC_ * 4), *c2 = (float*)alloc(B_ * DEC_ * 4);
  _Float16 *h_h  = (_Float16*)alloc(B_ * DEC_ * 2);
  _Float16 *h2_h = (_Float16*)alloc(B_ * DEC_ * 2);
  float* att2  = (float*)alloc(B_ * ATT_ * 4);
  float* catt2 = (float*)alloc(B_ * ATTC_ * 4);
  float* e_buf = (float*)alloc(B_ * P_ * 4);
  float* alpha = (float*)alloc(B_ * P_ * 4);
  float* awe   = (float*)alloc(B_ * ENC_ * 4);
  _Float16* awe_h = (_Float16*)alloc(B_ * ENC_ * 2);
  float* awe2  = (float*)alloc(B_ * ATT_ * 4);
  float* ec    = (float*)alloc(B_ * ATTC_ * 4);
  float* alpha_c = (float*)alloc(B_ * ATTC_ * 4);
  float* awce  = (float*)alloc(B_ * P_ * 4);
  float* g1 = (float*)alloc(B_ * 4 * DEC_ * 4);
  float* g2 = (float*)alloc(B_ * 4 * DEC_ * 4);
  _Float16* x1_h = (_Float16*)alloc((size_t)B_ * 2048 * 2);
  _Float16* x2_h = (_Float16*)alloc((size_t)B_ * K2PAD_ * 2);
  float* amf  = (float*)alloc(B_ * 4);
  float* scal = (float*)alloc(256);
  int* sort_ind = (int*)alloc(B_ * 4);
  int* dec_len  = (int*)alloc(B_ * 4);
  int* caps_s   = (int*)alloc(B_ * MAXLEN_ * 4);
  // f16 weight copies
  auto allocH = [&](size_t n) { return (_Float16*)alloc(n * 2); };
  _Float16* att_enc_Wh = allocH((size_t)ATT_ * ENC_);
  _Float16* chan_enc_Wh= allocH((size_t)ATTC_ * ENC_);
  _Float16* chan_out_Wh= allocH((size_t)ATTC_ * ENC_);
  _Float16* att_dec_Wh = allocH((size_t)ATT_ * DEC_);
  _Float16* chan_dec_Wh= allocH((size_t)ATTC_ * DEC_);
  _Float16* att_div_Wh = allocH((size_t)ATT_ * ENC_);
  _Float16* feat_div_Wh= allocH((size_t)ATT_ * ENC_);
  _Float16* init_h_Wh  = allocH((size_t)DEC_ * ENC_);
  _Float16* init_c_Wh  = allocH((size_t)DEC_ * ENC_);
  _Float16* l1_Wih_h = allocH((size_t)4 * DEC_ * 2048);
  _Float16* l1_Whh_h = allocH((size_t)4 * DEC_ * DEC_);
  _Float16* l2_Wih_h = allocH((size_t)4 * DEC_ * K2PAD_);  // padded
  _Float16* l2_Whh_h = allocH((size_t)4 * DEC_ * DEC_);
  _Float16* fc_Wh    = allocH((size_t)VOCAB_ * DEC_);

  auto castW = [&](const float* s, _Float16* d, size_t n) {
    k_cast<<<dim3((unsigned)((n + 255) / 256)), 256, 0, stream>>>(s, d, (int)n);
  };
  // -------- one-time setup --------
  castW(att_enc_W, att_enc_Wh, (size_t)ATT_ * ENC_);
  castW(chan_enc_W, chan_enc_Wh, (size_t)ATTC_ * ENC_);
  castW(chan_out_W, chan_out_Wh, (size_t)ATTC_ * ENC_);
  castW(att_dec_W, att_dec_Wh, (size_t)ATT_ * DEC_);
  castW(chan_dec_W, chan_dec_Wh, (size_t)ATTC_ * DEC_);
  castW(att_div_W, att_div_Wh, (size_t)ATT_ * ENC_);
  castW(feat_div_W, feat_div_Wh, (size_t)ATT_ * ENC_);
  castW(init_h_W, init_h_Wh, (size_t)DEC_ * ENC_);
  castW(init_c_W, init_c_Wh, (size_t)DEC_ * ENC_);
  castW(l1_Wih, l1_Wih_h, (size_t)4 * DEC_ * 2048);
  castW(l1_Whh, l1_Whh_h, (size_t)4 * DEC_ * DEC_);
  castW(l2_Whh, l2_Whh_h, (size_t)4 * DEC_ * DEC_);
  castW(fc_W, fc_Wh, (size_t)VOCAB_ * DEC_);
  k_cast_pad<<<(2048 * K2PAD_ + 255) / 256, 256, 0, stream>>>(l2_Wih, l2_Wih_h);

  k_sort<<<1, B_, 0, stream>>>(lens_in, caps_in, sort_ind, dec_len, caps_s,
                               outf + OUT_CAPS, outf + OUT_DECLEN, outf + OUT_SORTIND);
  k_gather_cast<<<B_ * P_, 256, 0, stream>>>(enc_in, sort_ind, enc_s, enc_h);
  k_mean<<<B_, 256, 0, stream>>>(enc_s, mean_f, mean_h);

  // init projections (M=64 WMMA GEMMs)
  gemm(stream, mean_h, feat_div_Wh, feat_div_b, img_gm, nullptr, B_, ATT_, ENC_, ATT_, 0);
  gemm(stream, mean_h, init_h_Wh, init_h_b, h, nullptr, B_, DEC_, ENC_, DEC_, 0);
  gemm(stream, mean_h, init_c_Wh, init_c_b, c, nullptr, B_, DEC_, ENC_, DEC_, 0);
  k_init_hc<<<(B_ * DEC_ + 255) / 256, 256, 0, stream>>>(h, c, h2, c2, h_h, h2_h);

  // big precompute GEMMs: (12544 x 2048) x (1024 x 2048)^T
  gemm(stream, enc_h, att_enc_Wh, att_enc_b, att1, nullptr, B_ * P_, ATT_, ENC_, ATT_, 0);
  gemm(stream, enc_h, chan_enc_Wh, chan_enc_b, catt1, nullptr, B_ * P_, ATTC_, ENC_, ATTC_, 0);
  gemm(stream, enc_h, chan_out_Wh, chan_out_b, imf, nullptr, B_ * P_, ATTC_, ENC_, ATTC_, 0);

  // -------- timestep loop --------
  for (int t = 0; t < T_; ++t) {
    k_active<<<1, 64, 0, stream>>>(dec_len, amf, t);
    gemm(stream, h_h, att_dec_Wh, att_dec_b, att2, nullptr, B_, ATT_, DEC_, ATT_, 0);
    gemm(stream, h_h, chan_dec_Wh, chan_dec_b, catt2, nullptr, B_, ATTC_, DEC_, ATTC_, 0);

    k_e_dot<<<B_ * P_, 256, 0, stream>>>(att1, att2, att_full_W, att_full_b, e_buf);
    k_softmax_p<<<B_, 256, 0, stream>>>(e_buf, alpha, outf + OUT_ALPHAS, amf, t);
    k_awe<<<dim3(B_, ENC_ / 256), 256, 0, stream>>>(enc_s, alpha, awe, awe_h);
    gemm(stream, awe_h, att_div_Wh, att_div_b, awe2, nullptr, B_, ATT_, ENC_, ATT_, 0);

    k_ec<<<(B_ * ATTC_ + 255) / 256, 256, 0, stream>>>(catt1, catt2, chan_full_W, chan_full_b, ec);
    k_softmax_a<<<B_, 256, 0, stream>>>(ec, alpha_c, outf + OUT_ALPHASC, amf, t);
    k_awce<<<B_ * P_, 256, 0, stream>>>(imf, alpha_c, awce);

    k_std_coff<<<1, 256, 0, stream>>>(awe2, awce, amf, scal);

    // LSTM 1
    k_build_x1<<<B_, 256, 0, stream>>>(embedding, caps_s, h2, img_gm, x1_h, t);
    gemm(stream, x1_h, l1_Wih_h, l1_bih, g1, nullptr, B_, 4 * DEC_, 2048, 4 * DEC_, 0);
    gemm(stream, h_h, l1_Whh_h, l1_bhh, g1, nullptr, B_, 4 * DEC_, DEC_, 4 * DEC_, 1);
    k_lstm<<<(B_ * DEC_ + 255) / 256, 256, 0, stream>>>(g1, h, c, h_h, amf);

    // LSTM 2
    k_build_x2<<<B_, 256, 0, stream>>>(awce, awe2, h, scal, x2_h);
    gemm(stream, x2_h, l2_Wih_h, l2_bih, g2, nullptr, B_, 4 * DEC_, K2PAD_, 4 * DEC_, 0);
    gemm(stream, h2_h, l2_Whh_h, l2_bhh, g2, nullptr, B_, 4 * DEC_, DEC_, 4 * DEC_, 1);
    k_lstm<<<(B_ * DEC_ + 255) / 256, 256, 0, stream>>>(g2, h2, c2, h2_h, amf);

    // vocab projection, masked, written straight into the output tensor
    gemm(stream, h2_h, fc_Wh, fc_b, outf + OUT_PRED + (size_t)t * VOCAB_, amf,
         B_, VOCAB_, DEC_, T_ * VOCAB_, 0);
  }
}
